// ROISelector_79663053406631
// MI455X (gfx1250) — matrix operand order repaired and checked by reading, not compiled, not run on hardware
//
#include <hip/hip_runtime.h>
#include <cstdint>

typedef __attribute__((ext_vector_type(16))) _Float16 v16h;
typedef __attribute__((ext_vector_type(8)))  float    v8f;

#define T_DIM 16
#define PIX   3136          // 56*56
#define HU    448
#define WU    448
#define NPIX  200704        // 448*448
#define SW    449           // integral image width
#define NSC   12
#define NB    8

__device__ __forceinline__ int reflect_idx(int i, int n) {
    if (i < 0) i = -i;
    if (i >= n) i = 2 * n - 2 - i;
    return i;
}

__device__ __forceinline__ void gauss_w(float* w, int k, float sigma) {
    float sum = 0.f;
    for (int j = 0; j < k; ++j) {
        float d = (float)j - (float)(k - 1) * 0.5f;
        w[j] = __expf(-(d * d) / (2.f * sigma * sigma));
        sum += w[j];
    }
    float inv = 1.f / sum;
    for (int j = 0; j < k; ++j) w[j] *= inv;
}

// ---------------------------------------------------------------------------
// Stage 1: channel mean + per-(t) min/max normalize.  x: (T,3,56,56) -> out (T,3136)
__global__ void k_mean_norm(const float* __restrict__ x, float* __restrict__ out) {
    const int t   = blockIdx.x;
    const int tid = threadIdx.x;
    const float* xt = x + (size_t)t * 3 * PIX;
    float* ot = out + (size_t)t * PIX;
    __shared__ float smn[256], smx[256];
    float mn = 1e30f, mx = -1e30f;
    for (int i = tid; i < PIX; i += 256) {
        float m = (xt[i] + xt[PIX + i] + xt[2 * PIX + i]) * (1.0f / 3.0f);
        ot[i] = m;
        mn = fminf(mn, m);
        mx = fmaxf(mx, m);
    }
    smn[tid] = mn; smx[tid] = mx;
    __syncthreads();
    for (int s = 128; s > 0; s >>= 1) {
        if (tid < s) {
            smn[tid] = fminf(smn[tid], smn[tid + s]);
            smx[tid] = fmaxf(smx[tid], smx[tid + s]);
        }
        __syncthreads();
    }
    mn = smn[0]; mx = smx[0];
    float scale = 1.0f / (mx - mn + 1e-5f);
    for (int i = tid; i < PIX; i += 256)
        ot[i] = (ot[i] - mn) * scale;
}

// ---------------------------------------------------------------------------
// Stage 2: 7-tap Gaussian along T (reflect), at base 56x56 resolution.
__global__ void k_tconv(const float* __restrict__ in, float* __restrict__ out) {
    const int pix = blockIdx.x * 256 + threadIdx.x;
    if (pix >= PIX) return;
    float w7[7];
    gauss_w(w7, 7, 1.75f);
    float v[T_DIM];
    for (int t = 0; t < T_DIM; ++t) v[t] = in[(size_t)t * PIX + pix];
    for (int t = 0; t < T_DIM; ++t) {
        float acc = 0.f;
        #pragma unroll
        for (int j = 0; j < 7; ++j)
            acc += w7[j] * v[reflect_idx(t - 3 + j, T_DIM)];
        out[(size_t)t * PIX + pix] = acc;
    }
}

// ---------------------------------------------------------------------------
// Stage 3: 8x nearest upsample, f32 -> f16.  (T,56,56) -> (T,448,448)
__global__ void k_upsample(const float* __restrict__ in, _Float16* __restrict__ out) {
    const int idx = blockIdx.x * 256 + threadIdx.x;   // T*NPIX total, exact grid
    const int t   = idx / NPIX;
    const int r   = idx % NPIX;
    const int y   = r / WU, xx = r % WU;
    out[idx] = (_Float16)in[(size_t)t * PIX + (y >> 3) * 56 + (xx >> 3)];
}

// ---------------------------------------------------------------------------
// Stage 4a: vertical 9-tap Gaussian (reflect) via WMMA.
// Per wave: one 16x16 tile.  A = banded weights (16x32), B = 32x16 input slab.
__global__ void k_convV(const _Float16* __restrict__ in, _Float16* __restrict__ out) {
    __shared__ _Float16 slabs[8][32 * 16];           // per-wave 32 rows x 16 cols
    const int wave = threadIdx.x >> 5;
    const int lane = threadIdx.x & 31;
    const int tile = blockIdx.x * 8 + wave;          // 16*28*28 tiles total
    const int t    = tile / 784;
    const int rr   = tile % 784;
    const int ty   = rr / 28, tx = rr % 28;
    const int row0 = ty * 16, col0 = tx * 16;
    const _Float16* base = in + (size_t)t * NPIX;
    _Float16* slab = slabs[wave];

    if (ty > 0 && ty < 27) {
        // interior: contiguous 32B per row, 16B aligned -> async copy to LDS
        const _Float16* rp = base + (size_t)(row0 - 4 + lane) * WU + col0;
        unsigned lds0 = (unsigned)(uintptr_t)(slab + lane * 16);
        unsigned long long ga = (unsigned long long)(uintptr_t)rp;
        asm volatile("global_load_async_to_lds_b128 %0, %1, off" :: "v"(lds0), "v"(ga) : "memory");
        asm volatile("global_load_async_to_lds_b128 %0, %1, off offset:16" :: "v"(lds0), "v"(ga) : "memory");
        asm volatile("s_wait_asynccnt 0x0" ::: "memory");
    } else {
        for (int i = lane; i < 512; i += 32) {
            int r = i >> 4, c = i & 15;
            slab[i] = base[(size_t)reflect_idx(row0 - 4 + r, HU) * WU + col0 + c];
        }
    }

    float w9[9];
    gauss_w(w9, 9, 2.25f);
    const int ln15 = lane & 15, g = lane >> 4;

    v16h A;                                          // A[m][k] = w9[k-m]
    #pragma unroll
    for (int e = 0; e < 16; ++e) {
        int K = g * 16 + e;
        int d = K - ln15;
        A[e] = (d >= 0 && d <= 8) ? (_Float16)w9[d] : (_Float16)0.f;
    }
    v16h Bm;                                         // B[k][n] = in[row0-4+k][col0+n]
    #pragma unroll
    for (int e = 0; e < 16; ++e) {
        int K = g * 16 + e;
        Bm[e] = slab[K * 16 + ln15];
    }
    v8f C = {};
    v8f D = __builtin_amdgcn_wmma_f32_16x16x32_f16(false, A, false, Bm, (short)0, C, false, false);

    _Float16* op = out + (size_t)t * NPIX;
    #pragma unroll
    for (int r8 = 0; r8 < 8; ++r8) {
        int mm = r8 + 8 * g;
        op[(size_t)(row0 + mm) * WU + col0 + ln15] = (_Float16)D[r8];
    }
}

// ---------------------------------------------------------------------------
// Stage 4b: horizontal 9-tap Gaussian (reflect) via WMMA.
// Per wave: one 16x16 tile.  A = 16x32 input slab, B = banded weights (32x16).
__global__ void k_convH(const _Float16* __restrict__ in, _Float16* __restrict__ out) {
    __shared__ _Float16 slabs[8][16 * 32];           // per-wave 16 rows x 32 cols
    const int wave = threadIdx.x >> 5;
    const int lane = threadIdx.x & 31;
    const int tile = blockIdx.x * 8 + wave;
    const int t    = tile / 784;
    const int rr   = tile % 784;
    const int ty   = rr / 28, tx = rr % 28;
    const int row0 = ty * 16, col0 = tx * 16;
    const int colStart = col0 - 8;                   // keeps 16B alignment
    const _Float16* base = in + (size_t)t * NPIX;
    _Float16* slab = slabs[wave];

    if (tx > 0 && tx < 27) {
        const int r = lane & 15, seg = lane >> 4;    // 16 rows x 2 segments of 32B
        const _Float16* rp = base + (size_t)(row0 + r) * WU + colStart + seg * 16;
        unsigned lds0 = (unsigned)(uintptr_t)(slab + r * 32 + seg * 16);
        unsigned long long ga = (unsigned long long)(uintptr_t)rp;
        asm volatile("global_load_async_to_lds_b128 %0, %1, off" :: "v"(lds0), "v"(ga) : "memory");
        asm volatile("global_load_async_to_lds_b128 %0, %1, off offset:16" :: "v"(lds0), "v"(ga) : "memory");
        asm volatile("s_wait_asynccnt 0x0" ::: "memory");
    } else {
        for (int i = lane; i < 512; i += 32) {
            int r = i >> 5, c = i & 31;
            slab[i] = base[(size_t)(row0 + r) * WU + reflect_idx(colStart + c, WU)];
        }
    }

    float w9[9];
    gauss_w(w9, 9, 2.25f);
    const int ln15 = lane & 15, g = lane >> 4;

    v16h A;                                          // A[m][k] = in[row0+m][col0-4+k]
    #pragma unroll
    for (int e = 0; e < 16; ++e) {
        int K = g * 16 + e;
        int ci = K + 4;                              // slab col; K>=24 has zero weight
        if (ci > 31) ci = 31;
        A[e] = slab[ln15 * 32 + ci];
    }
    v16h Bm;                                         // B[k][n] = w9[k-n]
    #pragma unroll
    for (int e = 0; e < 16; ++e) {
        int K = g * 16 + e;
        int d = K - ln15;
        Bm[e] = (d >= 0 && d <= 8) ? (_Float16)w9[d] : (_Float16)0.f;
    }
    v8f C = {};
    v8f D = __builtin_amdgcn_wmma_f32_16x16x32_f16(false, A, false, Bm, (short)0, C, false, false);

    _Float16* op = out + (size_t)t * NPIX;
    #pragma unroll
    for (int r8 = 0; r8 < 8; ++r8) {
        int mm = r8 + 8 * g;
        op[(size_t)(row0 + mm) * WU + col0 + ln15] = (_Float16)D[r8];
    }
}

// ---------------------------------------------------------------------------
// Stage 5: integral image.  S is (T,449,449), leading zero row/col.
__global__ void k_rowcum(const _Float16* __restrict__ g, float* __restrict__ S) {
    const int id = blockIdx.x * 256 + threadIdx.x;   // T*448
    if (id >= T_DIM * HU) return;
    const int t = id / HU, row = id % HU;
    const _Float16* gr = g + (size_t)t * NPIX + (size_t)row * WU;
    float* Sr = S + (size_t)t * SW * SW + (size_t)(row + 1) * SW;
    float acc = 0.f;
    Sr[0] = 0.f;
    for (int c = 0; c < WU; ++c) { acc += (float)gr[c]; Sr[c + 1] = acc; }
    if (row == 0) {
        float* S0 = S + (size_t)t * SW * SW;
        for (int c = 0; c < SW; ++c) S0[c] = 0.f;
    }
}

__global__ void k_colcum(float* __restrict__ S, float* __restrict__ denom) {
    const int id = blockIdx.x * 256 + threadIdx.x;   // T*448
    if (id >= T_DIM * HU) return;
    const int t = id / HU, col = id % HU + 1;
    float* St = S + (size_t)t * SW * SW;
    float acc = 0.f;
    for (int row = 1; row < SW; ++row) {
        acc += St[(size_t)row * SW + col];
        St[(size_t)row * SW + col] = acc;
    }
    if (col == HU) denom[t] = acc + 1e-5f;           // total sum + eps
}

// ---------------------------------------------------------------------------
// Stage 6: per (scale,t) box-filter max + first-argmax.  One block per (s,t).
__global__ void k_box(const float* __restrict__ S, float* __restrict__ vals,
                      int* __restrict__ idxs) {
    const int s = blockIdx.x >> 4;                   // 12 scales
    const int t = blockIdx.x & 15;
    const int ph = 56 + 14 * s;                      // half box (== pw)
    const float* St = S + (size_t)t * SW * SW;
    const int tid = threadIdx.x;
    float best = -1e30f;
    int bi = 0x7fffffff;
    for (int i = tid; i < NPIX; i += 1024) {
        int iy = i / WU, ix = i % WU;
        int y1 = max(iy - ph, 0), y2 = min(iy + ph + 1, HU);
        int x1 = max(ix - ph, 0), x2 = min(ix + ph + 1, WU);
        float v = St[(size_t)y2 * SW + x2] - St[(size_t)y1 * SW + x2]
                - St[(size_t)y2 * SW + x1] + St[(size_t)y1 * SW + x1];
        if (v > best || (v == best && i < bi)) { best = v; bi = i; }
    }
    __shared__ float sv[1024];
    __shared__ int   si[1024];
    sv[tid] = best; si[tid] = bi;
    __syncthreads();
    for (int k = 512; k > 0; k >>= 1) {
        if (tid < k) {
            if (sv[tid + k] > sv[tid] || (sv[tid + k] == sv[tid] && si[tid + k] < si[tid])) {
                sv[tid] = sv[tid + k];
                si[tid] = si[tid + k];
            }
        }
        __syncthreads();
    }
    if (tid == 0) { vals[blockIdx.x] = sv[0]; idxs[blockIdx.x] = si[0]; }
}

// ---------------------------------------------------------------------------
// Stage 7: threshold scan + output.  out = [centers (8,16,2), sizes (8,16,2)]
__global__ void k_final(const float* __restrict__ vals, const int* __restrict__ idxs,
                        const float* __restrict__ denom, float* __restrict__ out) {
    const int id = threadIdx.x;
    if (id >= NB * T_DIM) return;
    const int b = id >> 4, t = id & 15;
    const float dn = denom[b * T_DIM + t];
    int first = -1;
    for (int s = 0; s < NSC; ++s) {
        float v = vals[((size_t)b * NSC + s) * T_DIM + t] / dn;
        if (v > 0.3f) { first = s; break; }
    }
    float center, sh;
    if (first >= 0) {
        center = (float)idxs[((size_t)b * NSC + first) * T_DIM + t];
        sh = 112.f + 28.f * (float)first;
    } else {
        center = 0.f;
        sh = 420.f;
    }
    if (center == 0.f) center = 1568.f;              // H*W//2 quirk in reference
    float cy = floorf(center / 448.f) / 448.f;
    float cx = fmodf(center, 448.f) / 448.f;
    out[(size_t)(b * T_DIM + t) * 2 + 0] = cy;
    out[(size_t)(b * T_DIM + t) * 2 + 1] = cx;
    out[256 + (size_t)(b * T_DIM + t) * 2 + 0] = sh;
    out[256 + (size_t)(b * T_DIM + t) * 2 + 1] = sh;
}

// ---------------------------------------------------------------------------
extern "C" void kernel_launch(void* const* d_in, const int* in_sizes, int n_in,
                              void* d_out, int out_size, void* d_ws, size_t ws_size,
                              hipStream_t stream) {
    const float* x = (const float*)d_in[0];          // (8,16,3,56,56) f32
    float* out = (float*)d_out;                      // 512 f32
    char* ws = (char*)d_ws;

    // workspace layout (reused across batches), total ~26.2 MB
    float*    bufA  = (float*)(ws + 0);              // T*3136 f32
    float*    bufB  = (float*)(ws + 200704);         // T*3136 f32
    _Float16* bufC  = (_Float16*)(ws + 401408);      // T*448*448 f16
    _Float16* bufD  = (_Float16*)(ws + 6823936);     // T*448*448 f16
    float*    Simg  = (float*)(ws + 13246464);       // T*449*449 f32
    float*    vals  = (float*)(ws + 26148928);       // 8*12*16 f32
    int*      idxs  = (int*)(ws + 26155072);         // 8*12*16 i32
    float*    denom = (float*)(ws + 26161216);       // 8*16 f32

    for (int b = 0; b < NB; ++b) {
        const float* xb = x + (size_t)b * T_DIM * 3 * PIX;
        k_mean_norm<<<T_DIM, 256, 0, stream>>>(xb, bufA);
        k_tconv<<<13, 256, 0, stream>>>(bufA, bufB);     // T-conv pass 1
        k_tconv<<<13, 256, 0, stream>>>(bufB, bufA);     // T-conv pass 2
        k_upsample<<<12544, 256, 0, stream>>>(bufA, bufC);
        k_convV<<<1568, 256, 0, stream>>>(bufC, bufD);   // V pass 1 (WMMA)
        k_convH<<<1568, 256, 0, stream>>>(bufD, bufC);   // H pass 1 (WMMA)
        k_convV<<<1568, 256, 0, stream>>>(bufC, bufD);   // V pass 2 (WMMA)
        k_convH<<<1568, 256, 0, stream>>>(bufD, bufC);   // H pass 2 (WMMA)
        k_rowcum<<<28, 256, 0, stream>>>(bufC, Simg);
        k_colcum<<<28, 256, 0, stream>>>(Simg, denom + b * T_DIM);
        k_box<<<NSC * T_DIM, 1024, 0, stream>>>(Simg, vals + b * NSC * T_DIM,
                                                idxs + b * NSC * T_DIM);
    }
    k_final<<<1, 128, 0, stream>>>(vals, idxs, denom, out);
}
